// AttentionWithRelativePosition_78675210928276
// MI455X (gfx1250) — matrix operand order repaired and checked by reading, not compiled, not run on hardware
//
#include <hip/hip_runtime.h>
#include <hip/hip_bf16.h>

typedef __bf16 bhalf;
typedef __attribute__((ext_vector_type(16))) __bf16 v16bf;
typedef __attribute__((ext_vector_type(8)))  float   v8f;
typedef __attribute__((ext_vector_type(4)))  float   f32x4;
typedef __attribute__((ext_vector_type(4)))  unsigned int u32x4;
typedef int v4i __attribute__((vector_size(16)));   // matches builtin's <4 x i32>

union FragAB {
    v16bf v;
    u32x4 q[2];
    unsigned int u[8];
};

// ---------------------------------------------------------------------------
// gfx1250 async global->LDS copy (ASYNCcnt-tracked DMA) with safe fallback.
// Builtin signature (from hipcc diagnostics): first arg is
// <4 x i32> addrspace(1)* (global src), second is the LDS destination.
// ---------------------------------------------------------------------------
#if defined(__gfx1250__) && \
    __has_builtin(__builtin_amdgcn_global_load_async_to_lds_b128) && \
    __has_builtin(__builtin_amdgcn_s_wait_asynccnt)
#define ASYNC_LDS 1
#endif

#ifdef ASYNC_LDS
#define ASYNC_COPY16(dst_lds, src_glob)                                     \
    __builtin_amdgcn_global_load_async_to_lds_b128(                         \
        (__attribute__((address_space(1))) v4i*)(src_glob),                 \
        (__attribute__((address_space(3))) v4i*)(dst_lds), 0, 0)
#define ASYNC_WAIT() __builtin_amdgcn_s_wait_asynccnt(0)
#else
#define ASYNC_COPY16(dst_lds, src_glob)                                     \
    (*(u32x4*)(dst_lds) = *(const u32x4*)(src_glob))
#define ASYNC_WAIT() ((void)0)
#endif

// ---------------------------------------------------------------------------
// f32 -> bf16 convert, 8 elements/thread (2x b128 load, 1x b128 store)
// ---------------------------------------------------------------------------
__global__ __launch_bounds__(256) void cvt_f32_bf16_v8(
    const float* __restrict__ s, bhalf* __restrict__ d, long n8) {
    union { bhalf h[8]; u32x4 v; } pk;
    long i = (long)blockIdx.x * blockDim.x + threadIdx.x;
    long stride = (long)gridDim.x * blockDim.x;
    for (; i < n8; i += stride) {
        f32x4 a = *(const f32x4*)&s[i * 8];
        f32x4 b = *(const f32x4*)&s[i * 8 + 4];
        #pragma unroll
        for (int j = 0; j < 4; ++j) { pk.h[j] = (bhalf)a[j]; pk.h[j + 4] = (bhalf)b[j]; }
        *(u32x4*)&d[i * 8] = pk.v;
    }
}

// ---------------------------------------------------------------------------
// bf16 WMMA GEMM: C[M,N] = A[M,K] @ B[K,N] (+bias). 128x128 block tile,
// 256 threads = 8 waves arranged 4(M) x 2(N); each wave owns 32x64
// (8 C fragments, 8 WMMAs per K-step of 32). A tile row-major (k contig,
// async-copied straight to LDS); B tile transposed [n][k] so every
// fragment read is a contiguous aligned ds_load_b128.
// ---------------------------------------------------------------------------
template <bool BF16OUT>
__global__ __launch_bounds__(256) void gemm_wmma(
    const bhalf* __restrict__ A, const bhalf* __restrict__ B,
    void* __restrict__ Cp, const float* __restrict__ bias,
    int M, int N, int K, int lda, int ldb, int ldc)
{
    __shared__ __align__(16) bhalf As[128 * 32];
    __shared__ __align__(16) bhalf Bt[128 * 32];

    const int t    = threadIdx.x;
    const int wave = t >> 5;
    const int lane = t & 31;
    const int half = lane >> 4;
    const int l16  = lane & 15;
    const int wm   = wave >> 1;      // 0..3 -> 32 rows each
    const int wn   = wave & 1;       // 0..1 -> 64 cols each
    const int row0 = blockIdx.y * 128;
    const int col0 = blockIdx.x * 128;

    v8f acc[2][4] = {};

    for (int k0 = 0; k0 < K; k0 += 32) {
        __syncthreads();
        // A tile 128x32 halves: 512 16B chunks, 2/thread, direct to LDS
        #pragma unroll
        for (int c = 0; c < 2; ++c) {
            int chunk = t + c * 256;
            int r  = chunk >> 2;
            int ko = (chunk & 3) * 8;
            ASYNC_COPY16(&As[r * 32 + ko],
                         &A[(long)(row0 + r) * lda + k0 + ko]);
        }
        // B tile 32x128 -> transposed Bt[n][k]: 512 chunks, 2/thread
        #pragma unroll
        for (int c = 0; c < 2; ++c) {
            int chunk = t + c * 256;
            int k  = chunk >> 4;
            int nb = (chunk & 15) * 8;
            u32x4 vv = *(const u32x4*)&B[(long)(k0 + k) * ldb + col0 + nb];
            const bhalf* h = (const bhalf*)&vv;
            #pragma unroll
            for (int i = 0; i < 8; ++i) Bt[(nb + i) * 32 + k] = h[i];
        }
        if (k0 + 32 < K)  // global_prefetch for next A tile
            __builtin_prefetch(&A[(long)(row0 + (t >> 1)) * lda + k0 + 32], 0, 1);
        ASYNC_WAIT();
        __syncthreads();

        FragAB a[2], b[4];
        #pragma unroll
        for (int f = 0; f < 2; ++f) {
            int m = wm * 32 + f * 16 + l16;
            a[f].q[0] = *(const u32x4*)&As[m * 32 + half * 8];
            a[f].q[1] = *(const u32x4*)&As[m * 32 + 16 + half * 8];
        }
        #pragma unroll
        for (int f = 0; f < 4; ++f) {
            int n = wn * 64 + f * 16 + l16;
            b[f].q[0] = *(const u32x4*)&Bt[n * 32 + half * 8];
            b[f].q[1] = *(const u32x4*)&Bt[n * 32 + 16 + half * 8];
        }
        #pragma unroll
        for (int fm = 0; fm < 2; ++fm)
            #pragma unroll
            for (int fn = 0; fn < 4; ++fn)
                acc[fm][fn] = __builtin_amdgcn_wmma_f32_16x16x32_bf16(
                    false, a[fm].v, false, b[fn].v, (short)0, acc[fm][fn],
                    false, false);
    }

    #pragma unroll
    for (int fm = 0; fm < 2; ++fm)
        #pragma unroll
        for (int fn = 0; fn < 4; ++fn) {
            int col  = col0 + wn * 64 + fn * 16 + l16;
            float bv = bias ? bias[col] : 0.0f;
            #pragma unroll
            for (int r = 0; r < 8; ++r) {
                int row = row0 + wm * 32 + fm * 16 + r + half * 8;
                float v = acc[fm][fn][r] + bv;
                if (BF16OUT)
                    ((bhalf*)Cp)[(long)row * ldc + col] = (bhalf)v;
                else
                    ((float*)Cp)[(long)row * ldc + col] = v;
            }
        }
}

// ---------------------------------------------------------------------------
// Flash attention with relative-position bias.
// Block = (bt, head, 128-query tile), 256 threads = 8 waves; each wave owns
// 16 complete query rows -> softmax row reductions are intra-wave shuffles.
// K/V tiles of 64 keys staged in LDS, reused by all 128 query rows.
// ---------------------------------------------------------------------------
#define L_SEQ   512
#define DH      64
#define NHEADS  16
#define D_MODEL 1024

__global__ __launch_bounds__(256) void attn_wmma(
    const bhalf* __restrict__ qkv,     // [BT*L][3*D], q|k|v on last dim
    const float* __restrict__ rel_emb, // [255][16]
    bhalf* __restrict__ out)           // [BT*L][D]
{
    __shared__ __align__(16) bhalf qs[128 * 64];  // Q tile, row-major (d contig)
    __shared__ __align__(16) bhalf ks[64 * 64];   // K tile, row-major (d contig)
    __shared__ __align__(16) bhalf vt[64 * 64];   // V tile transposed [d][key]
    __shared__ __align__(16) bhalf ps[128 * 64];  // P tile, row-major (key contig)
    __shared__ float relh[255];

    const int t    = threadIdx.x;
    const int wave = t >> 5;
    const int lane = t & 31;
    const int half = lane >> 4;
    const int l16  = lane & 15;

    const int blk  = blockIdx.x;
    const int qt   = blk & 3;          // 128-row query tile (4 per seq)
    const int head = (blk >> 2) & 15;
    const int bt   = blk >> 6;

    const long rowbase = (long)bt * L_SEQ;
    const int  ldq = 3 * D_MODEL;
    const bhalf* qp = qkv + rowbase * ldq + head * DH;
    const bhalf* kp = qp + D_MODEL;
    const bhalf* vp = qp + 2 * D_MODEL;

    if (t < 255) relh[t] = rel_emb[t * NHEADS + head];

    // Q tile: 128 rows x 64 halves = 1024 16B chunks, 4/thread, direct to LDS
    #pragma unroll
    for (int c = 0; c < 4; ++c) {
        int chunk = t + c * 256;
        int r = chunk >> 3, doff = (chunk & 7) * 8;
        ASYNC_COPY16(&qs[r * 64 + doff],
                     &qp[(long)(qt * 128 + r) * ldq + doff]);
    }
    ASYNC_WAIT();
    __syncthreads();

    FragAB qa[2];
    {
        int m = wave * 16 + l16;
        #pragma unroll
        for (int c = 0; c < 2; ++c) {
            qa[c].q[0] = *(const u32x4*)&qs[m * 64 + c * 32 + half * 8];
            qa[c].q[1] = *(const u32x4*)&qs[m * 64 + c * 32 + 16 + half * 8];
        }
    }

    v8f acc[4] = {};
    float m_i[8], l_i[8];
    #pragma unroll
    for (int r = 0; r < 8; ++r) { m_i[r] = -1e30f; l_i[r] = 0.0f; }
    const float scale = 0.125f; // dh^-0.5

    for (int kt = 0; kt < 8; ++kt) {
        __syncthreads();
        // K tile (async, row-major) + V tile (VGPR path, transposed)
        #pragma unroll
        for (int c = 0; c < 2; ++c) {
            int chunk = t + c * 256;
            int r = chunk >> 3, doff = (chunk & 7) * 8;
            ASYNC_COPY16(&ks[r * 64 + doff],
                         &kp[(long)(kt * 64 + r) * ldq + doff]);
            u32x4 vv = *(const u32x4*)&vp[(long)(kt * 64 + r) * ldq + doff];
            const bhalf* h = (const bhalf*)&vv;
            #pragma unroll
            for (int i = 0; i < 8; ++i) vt[(doff + i) * 64 + r] = h[i];
        }
        ASYNC_WAIT();
        __syncthreads();

        // S = Q @ K^T : wave's 16 rows x 64 keys (4 frags x 2 WMMAs)
        v8f s[4];
        #pragma unroll
        for (int fn = 0; fn < 4; ++fn) {
            int n = fn * 16 + l16;
            v8f c0 = {};
            #pragma unroll
            for (int c = 0; c < 2; ++c) {
                FragAB b;
                b.q[0] = *(const u32x4*)&ks[n * 64 + c * 32 + half * 8];
                b.q[1] = *(const u32x4*)&ks[n * 64 + c * 32 + 16 + half * 8];
                c0 = __builtin_amdgcn_wmma_f32_16x16x32_bf16(
                    false, qa[c].v, false, b.v, (short)0, c0, false, false);
            }
            s[fn] = c0;
        }

        // scale + relative-position bias
        #pragma unroll
        for (int fn = 0; fn < 4; ++fn) {
            int kj = kt * 64 + fn * 16 + l16;
            #pragma unroll
            for (int r = 0; r < 8; ++r) {
                int qi  = qt * 128 + wave * 16 + r + half * 8;
                int rel = qi - kj;
                rel = rel < -127 ? -127 : (rel > 127 ? 127 : rel);
                s[fn][r] = s[fn][r] * scale + relh[rel + 127];
            }
        }

        // online softmax row statistics (rows live in 16-lane halves)
        float rm[8], rs[8], alpha[8];
        #pragma unroll
        for (int r = 0; r < 8; ++r) {
            float m = fmaxf(fmaxf(s[0][r], s[1][r]), fmaxf(s[2][r], s[3][r]));
            #pragma unroll
            for (int sh = 1; sh < 16; sh <<= 1) m = fmaxf(m, __shfl_xor(m, sh, 32));
            rm[r] = m;
        }
        #pragma unroll
        for (int r = 0; r < 8; ++r) {
            float mn = fmaxf(m_i[r], rm[r]);
            alpha[r] = __expf(m_i[r] - mn);
            m_i[r]   = mn;
            rs[r]    = 0.0f;
        }
        #pragma unroll
        for (int fn = 0; fn < 4; ++fn)
            #pragma unroll
            for (int r = 0; r < 8; ++r) {
                float p = __expf(s[fn][r] - m_i[r]);
                rs[r] += p;
                int row = wave * 16 + r + half * 8;
                ps[row * 64 + fn * 16 + l16] = (bhalf)p;
            }
        #pragma unroll
        for (int r = 0; r < 8; ++r) {
            float sum = rs[r];
            #pragma unroll
            for (int sh = 1; sh < 16; sh <<= 1) sum += __shfl_xor(sum, sh, 32);
            l_i[r] = l_i[r] * alpha[r] + sum;
        }
        #pragma unroll
        for (int fd = 0; fd < 4; ++fd)
            #pragma unroll
            for (int r = 0; r < 8; ++r) acc[fd][r] *= alpha[r];

        __builtin_amdgcn_wave_barrier(); // ps written/read by same wave; keep order

        // O += P @ V
        FragAB pa[2];
        {
            int m = wave * 16 + l16;
            #pragma unroll
            for (int c = 0; c < 2; ++c) {
                pa[c].q[0] = *(const u32x4*)&ps[m * 64 + c * 32 + half * 8];
                pa[c].q[1] = *(const u32x4*)&ps[m * 64 + c * 32 + 16 + half * 8];
            }
        }
        #pragma unroll
        for (int fd = 0; fd < 4; ++fd) {
            int n = fd * 16 + l16;
            #pragma unroll
            for (int c = 0; c < 2; ++c) {
                FragAB b;
                b.q[0] = *(const u32x4*)&vt[n * 64 + c * 32 + half * 8];
                b.q[1] = *(const u32x4*)&vt[n * 64 + c * 32 + 16 + half * 8];
                acc[fd] = __builtin_amdgcn_wmma_f32_16x16x32_bf16(
                    false, pa[c].v, false, b.v, (short)0, acc[fd], false, false);
            }
        }
    }

    // normalize and write bf16 attention output [BT*L][D]
    #pragma unroll
    for (int fd = 0; fd < 4; ++fd) {
        int d = fd * 16 + l16;
        #pragma unroll
        for (int r = 0; r < 8; ++r) {
            int l = qt * 128 + wave * 16 + r + half * 8;
            float v = acc[fd][r] / l_i[r];
            out[(rowbase + l) * (long)D_MODEL + head * DH + d] = (bhalf)v;
        }
    }
}

// ---------------------------------------------------------------------------
extern "C" void kernel_launch(void* const* d_in, const int* in_sizes, int n_in,
                              void* d_out, int out_size, void* d_ws, size_t ws_size,
                              hipStream_t stream)
{
    const float* x       = (const float*)d_in[0];
    const float* w_qkv   = (const float*)d_in[1];
    const float* rel_emb = (const float*)d_in[2];
    const float* w_out   = (const float*)d_in[3];
    const float* b_out   = (const float*)d_in[4];

    const long BT = 16, D = 1024;
    const long M  = BT * 512; // 8192 rows

    // workspace layout (~92.3 MB total)
    char* ws = (char*)d_ws;
    bhalf* xb    = (bhalf*)ws; ws += (size_t)M * D * 2;          // 16.78 MB
    bhalf* wqkvb = (bhalf*)ws; ws += (size_t)D * 3 * D * 2;      //  6.29 MB
    bhalf* woutb = (bhalf*)ws; ws += (size_t)D * D * 2;          //  2.10 MB
    bhalf* qkvb  = (bhalf*)ws; ws += (size_t)M * 3 * D * 2;      // 50.33 MB
    bhalf* aob   = (bhalf*)ws;                                   // 16.78 MB

    cvt_f32_bf16_v8<<<1024, 256, 0, stream>>>(x, xb, M * D / 8);
    cvt_f32_bf16_v8<<<1024, 256, 0, stream>>>(w_qkv, wqkvb, D * 3 * D / 8);
    cvt_f32_bf16_v8<<<512,  256, 0, stream>>>(w_out, woutb, D * D / 8);

    // qkv = x @ w_qkv  (bf16 out)
    dim3 g1((unsigned)(3 * D / 128), (unsigned)(M / 128));
    gemm_wmma<true><<<g1, 256, 0, stream>>>(xb, wqkvb, qkvb, nullptr,
        (int)M, (int)(3 * D), (int)D, (int)D, (int)(3 * D), (int)(3 * D));

    // flash attention with rel-pos bias
    attn_wmma<<<1024, 256, 0, stream>>>(qkvb, rel_emb, aob);

    // out = attn_out @ w_out + b_out  (f32 out)
    dim3 g2((unsigned)(D / 128), (unsigned)(M / 128));
    gemm_wmma<false><<<g2, 256, 0, stream>>>(aob, woutb, d_out, b_out,
        (int)M, (int)D, (int)D, (int)D, (int)D, (int)D);
}
